// GATLayer_55044300866301
// MI455X (gfx1250) — matrix operand order repaired and checked by reading, not compiled, not run on hardware
//
#include <hip/hip_runtime.h>
#include <hip/hip_bf16.h>
#include <math.h>

typedef __attribute__((ext_vector_type(16))) _Float16 v16h;
typedef __attribute__((ext_vector_type(8)))  float    v8f;
typedef __attribute__((ext_vector_type(2)))  _Float16 h2;
typedef __attribute__((ext_vector_type(4)))  unsigned int u32x4;
typedef __attribute__((ext_vector_type(8)))  int          i32x8;
typedef __attribute__((ext_vector_type(4)))  int          i32x4;

union V16  { v16h v; h2 p[8]; };
union U32H { unsigned u; _Float16 h[2]; };

#define N_ROWS 8192
#define D      128
#define LDSP   136   // padded pitch (halves): 128 data + 8 pad = TDM pad 4 DWORDs / 64 DWORDs
#define OPITCH 132   // k1 transpose-staging pitch (halves)

// A-operand (16-bit, 16x32) K pattern per ISA 7.12.2:
// lanes 0-15: VGPR0..3 -> K={0,1},{2,3},{4,5},{6,7}; VGPR4..7 -> K={16,17}..{22,23}
// lanes 16-31: +8
__device__ __forceinline__ int a_kbase(int v) {
    return (v < 4) ? (2 * v) : (16 + 2 * (v - 4));
}

// ---------------- Kernel 1: Wh = X@W (f16 WMMA, f32 accum); stores WhhT [d][j] f16;
// s = Wh.a1, t = Wh.a2
__global__ __launch_bounds__(256) void k1_wh(const float* __restrict__ X,
                                             const float* __restrict__ W,
                                             const float* __restrict__ A,
                                             _Float16* __restrict__ WhhT,
                                             float* __restrict__ S,
                                             float* __restrict__ T)
{
    __shared__ _Float16 lds_w[D * LDSP];      // W, f16, col-major [n][k]
    __shared__ _Float16 lds_o[D * OPITCH];    // transpose staging [d][i_local]
    const int tid = threadIdx.x;

    // stage W (f32 row-major [k][n]) -> LDS f16 col-major lds_w[n*LDSP + k]
    for (int it = 0; it < (D * D) / 256; ++it) {
        int idx = tid + 256 * it;
        int k = idx >> 7, n = idx & 127;
        lds_w[n * LDSP + k] = (_Float16)W[idx];
    }
    __syncthreads();

    const int wave = tid >> 5;
    const int lane = tid & 31;
    const int row  = lane & 15;
    const bool hi  = lane >= 16;
    const int i0   = (blockIdx.x * 8 + wave) * 16;

    v8f c[8];
    #pragma unroll
    for (int n = 0; n < 8; ++n) c[n] = (v8f)0.0f;

    for (int kc = 0; kc < 4; ++kc) {
        V16 a;
        const float* xr = X + (i0 + row) * D + kc * 32 + (hi ? 16 : 0);
        #pragma unroll
        for (int v = 0; v < 8; ++v) {
            float2 f = *(const float2*)(xr + 2 * v);
            h2 t2; t2.x = (_Float16)f.x; t2.y = (_Float16)f.y;
            a.p[v] = t2;
        }
        #pragma unroll
        for (int n = 0; n < 8; ++n) {
            V16 b;
            const _Float16* bp = lds_w + (16 * n + row) * LDSP + kc * 32 + (hi ? 16 : 0);
            #pragma unroll
            for (int v = 0; v < 8; ++v)
                b.p[v] = *(const h2*)(bp + 2 * v);
            c[n] = __builtin_amdgcn_wmma_f32_16x16x32_f16(false, a.v, false, b.v,
                                                          (short)0, c[n], false, false);
        }
    }

    // s_i = Wh_i . a1 , t_i = Wh_i . a2  (reduce within each 16-lane half)
    float a1v[8], a2v[8];
    #pragma unroll
    for (int n = 0; n < 8; ++n) {
        a1v[n] = A[16 * n + row];
        a2v[n] = A[128 + 16 * n + row];
    }
    #pragma unroll
    for (int v = 0; v < 8; ++v) {
        float ps = 0.f, pt = 0.f;
        #pragma unroll
        for (int n = 0; n < 8; ++n) { ps += c[n][v] * a1v[n]; pt += c[n][v] * a2v[n]; }
        #pragma unroll
        for (int off = 8; off >= 1; off >>= 1) {
            ps += __shfl_xor(ps, off, 32);
            pt += __shfl_xor(pt, off, 32);
        }
        if (row == 0) {
            int M = v + (hi ? 8 : 0);
            S[i0 + M] = ps;
            T[i0 + M] = pt;
        }
    }

    // block-level transpose through LDS -> coalesced WhhT stores
    #pragma unroll
    for (int n = 0; n < 8; ++n)
        #pragma unroll
        for (int v = 0; v < 8; ++v) {
            int M = v + (hi ? 8 : 0);
            lds_o[(16 * n + row) * OPITCH + wave * 16 + M] = (_Float16)c[n][v];
        }
    __syncthreads();

    unsigned* WT = (unsigned*)WhhT;
    const int iblk0 = blockIdx.x * 128;
    for (int it = 0; it < 32; ++it) {
        int idx = tid + 256 * it;
        int dd = idx >> 6, ip = idx & 63;
        U32H u;
        u.h[0] = lds_o[dd * OPITCH + 2 * ip];
        u.h[1] = lds_o[dd * OPITCH + 2 * ip + 1];
        WT[(size_t)dd * (N_ROWS / 2) + (iblk0 >> 1) + ip] = u.u;
    }
}

// ---------------- Kernel 1b: global max of t (LeakyReLU monotone => m_i = f(s_i + tmax))
__global__ __launch_bounds__(256) void k_tmax(const float* __restrict__ T,
                                              float* __restrict__ tmax)
{
    __shared__ float red[256];
    float m = -3.4e38f;
    for (int i = threadIdx.x; i < N_ROWS; i += 256) m = fmaxf(m, T[i]);
    red[threadIdx.x] = m;
    __syncthreads();
    for (int s = 128; s > 0; s >>= 1) {
        if (threadIdx.x < s) red[threadIdx.x] = fmaxf(red[threadIdx.x], red[threadIdx.x + s]);
        __syncthreads();
    }
    if (threadIdx.x == 0) *tmax = red[0];
}

// Issue one TDM load: WhhT[0:128][j0:j0+128] -> LDS [d][j] with padded pitch LDSP
__device__ __forceinline__ void tdm_issue(const _Float16* WhhT, int j0, unsigned lds_off)
{
    unsigned long long ga = (unsigned long long)WhhT + 2ull * (unsigned long long)j0;

    u32x4 g0;
    g0[0] = 1u;                                        // count=1, user descriptor
    g0[1] = lds_off;                                   // lds_addr (bytes)
    g0[2] = (unsigned)(ga & 0xFFFFFFFFull);            // global_addr[31:0]
    g0[3] = (unsigned)((ga >> 32) & 0x01FFFFFFull)     // global_addr[56:32]
          | (2u << 30);                                // type = 2 (image)

    i32x8 g1;
    g1[0] = (int)((1u << 16)                           // data_size = 1 -> 2 bytes
          |       (1u << 20)                           // pad_enable
          |       (5u << 22)                           // pad_interval: 64 DWORDs
          |       (3u << 25));                         // pad_amount: 4 DWORDs (8 halves)
    g1[1] = (int)((8192u & 0xFFFFu) << 16);            // tensor_dim0[15:0] = 8192
    g1[2] = (int)((8192u >> 16) | ((128u & 0xFFFFu) << 16)); // dim0 hi | tensor_dim1 lo
    g1[3] = (int)((128u >> 16) | (128u << 16));        // dim1 hi | tile_dim0 = 128
    g1[4] = (int)(128u);                               // tile_dim1 = 128, tile_dim2 = 0
    g1[5] = (int)(8192u);                              // tensor_dim0_stride[31:0] = 8192
    g1[6] = 0;                                         // stride hi | tensor_dim1_stride lo
    g1[7] = 0;

    i32x4 z4 = {0, 0, 0, 0};                           // groups 2/3 unused (2D tensor)
    i32x8 z8 = {0, 0, 0, 0, 0, 0, 0, 0};
    __builtin_amdgcn_tensor_load_to_lds(g0, g1, z4, z4, z8, 0);
}

// ---------------- Kernel 2: fused softmax(rank-1 scores) + dropout + attn@Wh + ELU
// TDM double-buffered staging of Wh chunks.
__global__ __launch_bounds__(256) void k2_attn(const _Float16* __restrict__ WhhT,
                                               const float* __restrict__ S,
                                               const float* __restrict__ T,
                                               const float* __restrict__ tmaxp,
                                               float* __restrict__ Out)
{
    __shared__ _Float16 lds_wh[2][D * LDSP];   // double-buffered [d][j] chunks
    __shared__ float lds_t[2][128];
    __shared__ float lds_z[128];

    const int tid  = threadIdx.x;
    const int wave = tid >> 5, lane = tid & 31;
    const int row  = lane & 15;
    const bool hi  = lane >= 16;
    const int i0   = blockIdx.x * 128 + wave * 16;
    const int irow = i0 + row;

    const float s_row = S[irow];
    const float tmax  = *tmaxp;
    float em = s_row + tmax;
    const float m_row = (em > 0.f) ? em : 0.2f * em;

    v8f c[8];
    #pragma unroll
    for (int n = 0; n < 8; ++n) c[n] = (v8f)0.0f;
    float zsum = 0.f;

    const unsigned lds_off0 = (unsigned)(unsigned long long)(&lds_wh[0][0]);
    const unsigned lds_off1 = (unsigned)(unsigned long long)(&lds_wh[1][0]);

    // prologue: stage chunk 0
    if (tid < 128) lds_t[0][tid] = T[tid];
    if (wave == 0) tdm_issue(WhhT, 0, lds_off0);

    const int NCHUNK = N_ROWS / 128;
    for (int cix = 0; cix < NCHUNK; ++cix) {
        const int cur = cix & 1;

        if (cix + 1 < NCHUNK) {
            if (tid < 128) lds_t[1 - cur][tid] = T[(cix + 1) * 128 + tid];
            if (wave == 0) tdm_issue(WhhT, (cix + 1) * 128, cur ? lds_off0 : lds_off1);
            if (wave == 0) __builtin_amdgcn_s_wait_tensorcnt(1);
        } else {
            if (wave == 0) __builtin_amdgcn_s_wait_tensorcnt(0);
        }
        __syncthreads();   // chunk cix now fully in LDS for all waves

        const int j0 = cix * 128;
        const _Float16* buf = &lds_wh[cur][0];
        const float* tt = &lds_t[cur][0];

        for (int kc = 0; kc < 4; ++kc) {
            // build 16x32 probability tile (A operand) in registers
            V16 a;
            #pragma unroll
            for (int v = 0; v < 8; ++v) {
                int k0 = kc * 32 + 2 * v + (hi ? 16 : 0);
                float pd[2];
                #pragma unroll
                for (int q = 0; q < 2; ++q) {
                    int k = k0 + q;
                    float tj = tt[k];
                    float e  = s_row + tj;
                    float f  = (e > 0.f) ? e : 0.2f * e;
                    float p  = __expf(f - m_row);
                    zsum += p;                       // Z uses pre-dropout prob
                    unsigned hsh = (unsigned)irow * 8192u + (unsigned)(j0 + k);
                    hsh ^= hsh >> 16; hsh *= 0x7feb352du;
                    hsh ^= hsh >> 15; hsh *= 0x846ca68bu;
                    hsh ^= hsh >> 16;
                    pd[q] = (hsh < 3865470566u) ? p * (1.0f / 0.9f) : 0.0f; // keep=0.9
                }
                h2 t2; t2.x = (_Float16)pd[0]; t2.y = (_Float16)pd[1];
                a.p[v] = t2;
            }
            #pragma unroll
            for (int n = 0; n < 8; ++n) {
                V16 b;
                const _Float16* bp = buf + (16 * n + row) * LDSP + kc * 32 + (hi ? 16 : 0);
                #pragma unroll
                for (int v = 0; v < 8; ++v)
                    b.p[v] = *(const h2*)(bp + 2 * v);
                c[n] = __builtin_amdgcn_wmma_f32_16x16x32_f16(false, a.v, false, b.v,
                                                              (short)0, c[n], false, false);
            }
        }
        __syncthreads();   // all waves done with buf[cur] before next TDM overwrites it
    }

    // Z per row: each lane covered half the j's for its row; pair lanes L and L^16
    float z_total = zsum + __shfl_xor(zsum, 16, 32);
    if (!hi) lds_z[wave * 16 + row] = z_total;
    __syncthreads();

    #pragma unroll
    for (int v = 0; v < 8; ++v) {
        int M = v + (hi ? 8 : 0);
        float zr = 1.0f / lds_z[wave * 16 + M];
        #pragma unroll
        for (int n = 0; n < 8; ++n) {
            float val = c[n][v] * zr;
            float o = (val > 0.f) ? val : (__expf(val) - 1.0f);   // ELU
            Out[(size_t)(i0 + M) * D + 16 * n + row] = o;
        }
    }
}

extern "C" void kernel_launch(void* const* d_in, const int* in_sizes, int n_in,
                              void* d_out, int out_size, void* d_ws, size_t ws_size,
                              hipStream_t stream)
{
    const float* X = (const float*)d_in[0];   // [8192,128]
    const float* W = (const float*)d_in[1];   // [128,128]
    const float* A = (const float*)d_in[2];   // [256,1]
    float* Out = (float*)d_out;               // [8192,128]

    char* ws = (char*)d_ws;
    _Float16* WhhT = (_Float16*)ws;                                 // 2 MB, [128][8192]
    float* S    = (float*)(ws + 2u * 1024u * 1024u);                // 32 KB
    float* T    = (float*)(ws + 2u * 1024u * 1024u + 32u * 1024u);  // 32 KB
    float* tmax = (float*)(ws + 2u * 1024u * 1024u + 64u * 1024u);  // 4 B

    k1_wh<<<64, 256, 0, stream>>>(X, W, A, WhhT, S, T);
    k_tmax<<<1, 256, 0, stream>>>(T, tmax);
    k2_attn<<<64, 256, 0, stream>>>(WhhT, S, T, tmax, Out);
}
